// SparseGPT_61194694033949
// MI455X (gfx1250) — matrix operand order repaired and checked by reading, not compiled, not run on hardware
//
#include <hip/hip_runtime.h>
#include <math.h>

// ---- model dims -------------------------------------------------------------
#define TOK    2048      // B*T rows
#define BATCH  4
#define TSEQ   512
#define DIM    768
#define HEADS  12
#define HD     64
#define FFDIM  3072
#define VOCAB  50257
#define LAYERS 4
#define KW1    614       // round(0.2*FF)
#define KW2    154       // round(0.2*D)

// ---- WMMA types (CDNA5 / gfx1250, wave32) ----------------------------------
typedef __attribute__((ext_vector_type(16))) __bf16 v16bf;
typedef __attribute__((ext_vector_type(8)))  float  v8f;

__device__ __forceinline__ unsigned short f2bf(float f) {
  union { float f; unsigned u; } c; c.f = f;
  unsigned u = c.u;
  return (unsigned short)((u + 0x7FFFu + ((u >> 16) & 1u)) >> 16);  // RNE
}

// ---- gfx1250 async global->LDS (ASYNCcnt) with safe fallback ---------------
#if __has_builtin(__builtin_amdgcn_global_load_async_to_lds_b128)
#define HAS_ASYNC_LDS 1
#else
#define HAS_ASYNC_LDS 0
#endif

typedef int v4i32 __attribute__((vector_size(16)));
typedef __attribute__((address_space(1))) v4i32* gptr_v4;  // global v4i*
typedef __attribute__((address_space(3))) v4i32* lptr_v4;  // LDS v4i*

__device__ __forceinline__ void cp16_g2l(const unsigned short* g, unsigned short* l) {
#if HAS_ASYNC_LDS
  // flat LDS pointer: addr[31:0] is the LDS byte address (ISA aperture rule)
  __builtin_amdgcn_global_load_async_to_lds_b128(
      (gptr_v4)(size_t)g, (lptr_v4)(unsigned)(size_t)l, 0, 0);
#else
  *(uint4*)l = *(const uint4*)g;
#endif
}

__device__ __forceinline__ void wait_async_all() {
#if HAS_ASYNC_LDS
#if __has_builtin(__builtin_amdgcn_s_wait_asynccnt)
  __builtin_amdgcn_s_wait_asynccnt(0);
#else
  asm volatile("s_wait_asynccnt 0x0" ::: "memory");
#endif
#endif
}

// ---- fp32 -> bf16 bulk converter (vectorized) ------------------------------
__global__ __launch_bounds__(256)
void cvt_f32_to_bf16(const float* __restrict__ in, unsigned short* __restrict__ out,
                     int n4) {
  int i = blockIdx.x * 256 + threadIdx.x;
  if (i >= n4) return;
  float4 v = ((const float4*)in)[i];
  unsigned lo = (unsigned)f2bf(v.x) | ((unsigned)f2bf(v.y) << 16);
  unsigned hi = (unsigned)f2bf(v.z) | ((unsigned)f2bf(v.w) << 16);
  ((uint2*)out)[i] = make_uint2(lo, hi);
}

// ---- embedding gather ------------------------------------------------------
__global__ __launch_bounds__(256)
void embed_kernel(const int* __restrict__ idx, const float* __restrict__ tok,
                  const float* __restrict__ pos, float* __restrict__ x) {
  int row = blockIdx.x;
  int t   = row % TSEQ;
  int id  = idx[row];
  for (int c = threadIdx.x; c < DIM; c += 256)
    x[(size_t)row * DIM + c] = tok[(size_t)id * DIM + c] + pos[t * DIM + c];
}

// ---- layernorm over D=768, one block per row; optional bf16 shadow out -----
__global__ __launch_bounds__(256)
void layernorm_kernel(const float* __restrict__ in, const float* __restrict__ s,
                      const float* __restrict__ b, float* __restrict__ out,
                      unsigned short* __restrict__ out_bf) {
  __shared__ float red[256];
  int row = blockIdx.x, tid = threadIdx.x;
  const float* x = in + (size_t)row * DIM;
  float v0 = x[tid], v1 = x[tid + 256], v2 = x[tid + 512];
  red[tid] = v0 + v1 + v2;
  __syncthreads();
  for (int s2 = 128; s2 > 0; s2 >>= 1) {
    if (tid < s2) red[tid] += red[tid + s2];
    __syncthreads();
  }
  float mu = red[0] * (1.0f / DIM);
  __syncthreads();
  float d0 = v0 - mu, d1 = v1 - mu, d2 = v2 - mu;
  red[tid] = d0 * d0 + d1 * d1 + d2 * d2;
  __syncthreads();
  for (int s2 = 128; s2 > 0; s2 >>= 1) {
    if (tid < s2) red[tid] += red[tid + s2];
    __syncthreads();
  }
  float rs = rsqrtf(red[0] * (1.0f / DIM) + 1e-5f);
  float o0 = d0 * rs * s[tid]       + b[tid];
  float o1 = d1 * rs * s[tid + 256] + b[tid + 256];
  float o2 = d2 * rs * s[tid + 512] + b[tid + 512];
  float* o = out + (size_t)row * DIM;
  o[tid] = o0; o[tid + 256] = o1; o[tid + 512] = o2;
  if (out_bf) {
    unsigned short* ob = out_bf + (size_t)row * DIM;
    ob[tid] = f2bf(o0); ob[tid + 256] = f2bf(o1); ob[tid + 512] = f2bf(o2);
  }
}

// ============================================================================
// GEMM v2: A[M,K] bf16, W[N,K] bf16 -> out fp32. Async LDS staging, double
// buffered; 128x64 block tile, 8 waves x (32x32), v_wmma_f32_16x16x32_bf16.
// ============================================================================
#define MBLK 128
#define NBLK 64
#define KSTEP 32
#define LDA 40   // padded bf16 row stride; 80B rows keep 16B segment alignment
#define LDB 40

__global__ __launch_bounds__(256)
void gemm_bf16x2_wmma(const unsigned short* __restrict__ A,
                      const unsigned short* __restrict__ W,
                      const float* __restrict__ bias, const float* __restrict__ residual,
                      float* __restrict__ out, int M, int N, int K) {
  __shared__ unsigned short As[2][MBLK * LDA];
  __shared__ unsigned short Bs[2][NBLK * LDB];

  const int tid  = threadIdx.x;
  const int lane = tid & 31;
  const int wave = tid >> 5;
  const int wm   = wave >> 1;
  const int wn   = wave & 1;
  const int m0   = blockIdx.x * MBLK;
  const int n0   = blockIdx.y * NBLK;

  v8f acc[2][2];
#pragma unroll
  for (int i = 0; i < 2; ++i)
#pragma unroll
    for (int j = 0; j < 2; ++j) acc[i][j] = (v8f){0.f, 0.f, 0.f, 0.f, 0.f, 0.f, 0.f, 0.f};

  const int nk = K / KSTEP;

  // tile stagers: 16B async transfers, 4 segments per 64B row
  auto stageA = [&](int kb, int buf) {
#pragma unroll
    for (int it = 0; it < 2; ++it) {
      int e = tid + it * 256;        // 512 transfers
      int r = e >> 2, seg = e & 3;
      cp16_g2l(A + (size_t)(m0 + r) * K + kb + seg * 8,
               &As[buf][r * LDA + seg * 8]);
    }
  };
  auto stageB = [&](int kb, int buf) {
    int r = tid >> 2, seg = tid & 3;  // 256 transfers
    int n = n0 + r;
    if (n < N) {
      cp16_g2l(W + (size_t)n * K + kb + seg * 8, &Bs[buf][r * LDB + seg * 8]);
    } else {
      *(uint4*)&Bs[buf][r * LDB + seg * 8] = make_uint4(0u, 0u, 0u, 0u);
    }
  };

  stageA(0, 0);
  stageB(0, 0);

  for (int i = 0; i < nk; ++i) {
    wait_async_all();
    __syncthreads();                      // tile i ready on all waves
    if (i + 1 < nk) {                     // prefetch tile i+1 into other buffer
      stageA((i + 1) * KSTEP, (i + 1) & 1);
      stageB((i + 1) * KSTEP, (i + 1) & 1);
    }
    const unsigned short* Asb = &As[i & 1][0];
    const unsigned short* Bsb = &Bs[i & 1][0];

    union { v16bf v; unsigned d[8]; } af[2], bfg[2];
    const int kb8  = (lane >> 4) * 8;
    const int kb16 = (lane >> 4) * 16;
#pragma unroll
    for (int mi = 0; mi < 2; ++mi) {
      int row = wm * 32 + mi * 16 + (lane & 15);
#pragma unroll
      for (int j = 0; j < 8; ++j) {
        int kk = ((j >> 2) << 4) + kb8 + ((j & 3) << 1);
        af[mi].d[j] = *(const unsigned*)&Asb[row * LDA + kk];
      }
    }
#pragma unroll
    for (int ni = 0; ni < 2; ++ni) {
      int row = wn * 32 + ni * 16 + (lane & 15);
#pragma unroll
      for (int j = 0; j < 8; ++j)
        bfg[ni].d[j] = *(const unsigned*)&Bsb[row * LDB + ((j >> 2) << 4) + kb8 + ((j & 3) << 1)];
    }
#pragma unroll
    for (int mi = 0; mi < 2; ++mi)
#pragma unroll
      for (int ni = 0; ni < 2; ++ni)
        acc[mi][ni] = __builtin_amdgcn_wmma_f32_16x16x32_bf16(
            false, af[mi].v, false, bfg[ni].v, (short)0, acc[mi][ni], false, false);
  }

#pragma unroll
  for (int mi = 0; mi < 2; ++mi) {
#pragma unroll
    for (int ni = 0; ni < 2; ++ni) {
      int n = n0 + wn * 32 + ni * 16 + (lane & 15);
      if (n >= N) continue;
      float bb = bias ? bias[n] : 0.0f;
#pragma unroll
      for (int r = 0; r < 8; ++r) {
        int m = m0 + wm * 32 + mi * 16 + r + ((lane >> 4) << 3);
        size_t o = (size_t)m * N + n;
        float v = acc[mi][ni][r] + bb;
        if (residual) v += residual[o];
        out[o] = v;
      }
    }
  }
}

// ============================================================================
// GEMM v1 (fallback when workspace is small): A fp32, W fp32, convert in LDS
// ============================================================================
__global__ __launch_bounds__(256)
void gemm_bf16_wmma(const float* __restrict__ A, const float* __restrict__ W,
                    const float* __restrict__ bias, const float* __restrict__ residual,
                    float* __restrict__ out, int M, int N, int K) {
  __shared__ unsigned short As[MBLK * LDA];
  __shared__ unsigned short Bs[NBLK * LDB];

  const int tid  = threadIdx.x;
  const int lane = tid & 31;
  const int wave = tid >> 5;
  const int wm   = wave >> 1;
  const int wn   = wave & 1;
  const int m0   = blockIdx.x * MBLK;
  const int n0   = blockIdx.y * NBLK;

  v8f acc[2][2];
#pragma unroll
  for (int i = 0; i < 2; ++i)
#pragma unroll
    for (int j = 0; j < 2; ++j) acc[i][j] = (v8f){0.f, 0.f, 0.f, 0.f, 0.f, 0.f, 0.f, 0.f};

  for (int kb = 0; kb < K; kb += KSTEP) {
    // A tile via float4, pack 2 bf16 per dword
#pragma unroll
    for (int it = 0; it < 4; ++it) {
      int e = tid + it * 256;          // 1024 x float4
      int r = e >> 3, c4 = (e & 7) * 4;
      float4 v = *(const float4*)&A[(size_t)(m0 + r) * K + kb + c4];
      unsigned lo = (unsigned)f2bf(v.x) | ((unsigned)f2bf(v.y) << 16);
      unsigned hi = (unsigned)f2bf(v.z) | ((unsigned)f2bf(v.w) << 16);
      *(uint2*)&As[r * LDA + c4] = make_uint2(lo, hi);
    }
#pragma unroll
    for (int it = 0; it < 2; ++it) {
      int e = tid + it * 256;          // 512 x float4
      int r = e >> 3, c4 = (e & 7) * 4;
      int n = n0 + r;
      uint2 pk = make_uint2(0u, 0u);
      if (n < N) {
        float4 v = *(const float4*)&W[(size_t)n * K + kb + c4];
        pk.x = (unsigned)f2bf(v.x) | ((unsigned)f2bf(v.y) << 16);
        pk.y = (unsigned)f2bf(v.z) | ((unsigned)f2bf(v.w) << 16);
      }
      *(uint2*)&Bs[r * LDB + c4] = pk;
    }
    __syncthreads();

    union { v16bf v; unsigned d[8]; } af[2], bfg[2];
    const int kb8  = (lane >> 4) * 8;
    const int kb16 = (lane >> 4) * 16;
#pragma unroll
    for (int mi = 0; mi < 2; ++mi) {
      int row = wm * 32 + mi * 16 + (lane & 15);
#pragma unroll
      for (int j = 0; j < 8; ++j) {
        int kk = ((j >> 2) << 4) + kb8 + ((j & 3) << 1);
        af[mi].d[j] = *(const unsigned*)&As[row * LDA + kk];
      }
    }
#pragma unroll
    for (int ni = 0; ni < 2; ++ni) {
      int row = wn * 32 + ni * 16 + (lane & 15);
#pragma unroll
      for (int j = 0; j < 8; ++j)
        bfg[ni].d[j] = *(const unsigned*)&Bs[row * LDB + ((j >> 2) << 4) + kb8 + ((j & 3) << 1)];
    }
#pragma unroll
    for (int mi = 0; mi < 2; ++mi)
#pragma unroll
      for (int ni = 0; ni < 2; ++ni)
        acc[mi][ni] = __builtin_amdgcn_wmma_f32_16x16x32_bf16(
            false, af[mi].v, false, bfg[ni].v, (short)0, acc[mi][ni], false, false);
    __syncthreads();
  }

#pragma unroll
  for (int mi = 0; mi < 2; ++mi) {
#pragma unroll
    for (int ni = 0; ni < 2; ++ni) {
      int n = n0 + wn * 32 + ni * 16 + (lane & 15);
      if (n >= N) continue;
      float bb = bias ? bias[n] : 0.0f;
#pragma unroll
      for (int r = 0; r < 8; ++r) {
        int m = m0 + wm * 32 + mi * 16 + r + ((lane >> 4) << 3);
        size_t o = (size_t)m * N + n;
        float v = acc[mi][ni][r] + bb;
        if (residual) v += residual[o];
        out[o] = v;
      }
    }
  }
}

// ---- causal attention, one wave per (b, h, q-row); optional bf16 shadow ----
__global__ __launch_bounds__(32)
void attention_kernel(const float* __restrict__ q, const float* __restrict__ k,
                      const float* __restrict__ v, float* __restrict__ y,
                      unsigned short* __restrict__ y_bf) {
  __shared__ float qs[HD];
  __shared__ float p[TSEQ];
  __shared__ float red[32];
  int blk  = blockIdx.x;
  int qt   = blk % TSEQ;
  int h    = (blk / TSEQ) % HEADS;
  int b    = blk / (TSEQ * HEADS);
  int lane = threadIdx.x;
  size_t base = ((size_t)b * TSEQ) * DIM + (size_t)h * HD;

  qs[lane]      = q[base + (size_t)qt * DIM + lane];
  qs[lane + 32] = q[base + (size_t)qt * DIM + lane + 32];
  __syncthreads();

  const float scale = 0.125f;  // 1/sqrt(64)
  float mx = -1e30f;
  for (int kk = lane; kk < TSEQ; kk += 32) {
    float sc = -1e30f;
    if (kk <= qt) {
      const float* kr = k + base + (size_t)kk * DIM;
      float d = 0.f;
#pragma unroll 8
      for (int c = 0; c < HD; ++c) d += qs[c] * kr[c];
      sc = d * scale;
    }
    p[kk] = sc;
    mx = fmaxf(mx, sc);
  }
  red[lane] = mx; __syncthreads();
  for (int s = 16; s > 0; s >>= 1) { if (lane < s) red[lane] = fmaxf(red[lane], red[lane + s]); __syncthreads(); }
  mx = red[0]; __syncthreads();

  float sum = 0.f;
  for (int kk = lane; kk < TSEQ; kk += 32) {
    float e = (kk <= qt) ? expf(p[kk] - mx) : 0.0f;
    p[kk] = e; sum += e;
  }
  red[lane] = sum; __syncthreads();
  for (int s = 16; s > 0; s >>= 1) { if (lane < s) red[lane] += red[lane + s]; __syncthreads(); }
  float inv = 1.0f / red[0];
  __syncthreads();

  float a0 = 0.f, a1 = 0.f;
  for (int kk = 0; kk <= qt; ++kk) {
    const float* vr = v + base + (size_t)kk * DIM;
    float pw = p[kk];
    a0 += pw * vr[lane];
    a1 += pw * vr[lane + 32];
  }
  a0 *= inv; a1 *= inv;
  y[base + (size_t)qt * DIM + lane]      = a0;
  y[base + (size_t)qt * DIM + lane + 32] = a1;
  if (y_bf) {
    y_bf[base + (size_t)qt * DIM + lane]      = f2bf(a0);
    y_bf[base + (size_t)qt * DIM + lane + 32] = f2bf(a1);
  }
}

// ---- exact KWinners (radix select); optional GELU + bf16 shadow ------------
__global__ __launch_bounds__(256)
void kwinners_kernel(float* __restrict__ z, int F, int k, int apply_gelu,
                     unsigned short* __restrict__ out_bf) {
  __shared__ int cnt;
  int row = blockIdx.x, tid = threadIdx.x;
  float* zr = z + (size_t)row * F;
  const int PT = F / 256;  // 12 (FF) or 3 (D)
  float zv[12]; unsigned key[12];
  for (int i = 0; i < PT; ++i) {
    float f = zr[tid + i * 256];
    zv[i] = f;
    union { float f; unsigned u; } c; c.f = f;
    key[i] = (c.u & 0x80000000u) ? ~c.u : (c.u | 0x80000000u);
  }
  unsigned thr = 0;
  for (int bit = 31; bit >= 0; --bit) {
    unsigned cand = thr | (1u << bit);
    if (tid == 0) cnt = 0;
    __syncthreads();
    int c = 0;
    for (int i = 0; i < PT; ++i) c += (key[i] >= cand) ? 1 : 0;
    atomicAdd(&cnt, c);
    __syncthreads();
    int total = cnt;
    __syncthreads();
    if (total >= k) thr = cand;
  }
  unsigned short* ob = out_bf ? out_bf + (size_t)row * F : nullptr;
  for (int i = 0; i < PT; ++i) {
    float f = (key[i] >= thr) ? zv[i] : 0.0f;
    if (apply_gelu) f = 0.5f * f * (1.0f + erff(f * 0.70710678118654752f));
    zr[tid + i * 256] = f;
    if (ob) ob[tid + i * 256] = f2bf(f);
  }
}

// ---- x += z ----------------------------------------------------------------
__global__ __launch_bounds__(256)
void add_kernel(float* __restrict__ x, const float* __restrict__ z, int n) {
  int i = blockIdx.x * 256 + threadIdx.x;
  if (i < n) x[i] += z[i];
}

// ---- driver ----------------------------------------------------------------
extern "C" void kernel_launch(void* const* d_in, const int* in_sizes, int n_in,
                              void* d_out, int out_size, void* d_ws, size_t ws_size,
                              hipStream_t stream) {
  const int*   idx  = (const int*)  d_in[0];
  const float* tok  = (const float*)d_in[1];
  const float* pos  = (const float*)d_in[2];
  const float* ln1s = (const float*)d_in[3];
  const float* ln1b = (const float*)d_in[4];
  const float* Wq   = (const float*)d_in[5];
  const float* bq   = (const float*)d_in[6];
  const float* Wk   = (const float*)d_in[7];
  const float* bk   = (const float*)d_in[8];
  const float* Wv   = (const float*)d_in[9];
  const float* bv   = (const float*)d_in[10];
  const float* Wp   = (const float*)d_in[11];
  const float* bp   = (const float*)d_in[12];
  const float* ln2s = (const float*)d_in[13];
  const float* ln2b = (const float*)d_in[14];
  const float* W1   = (const float*)d_in[15];
  const float* b1   = (const float*)d_in[16];
  const float* W2   = (const float*)d_in[17];
  const float* b2   = (const float*)d_in[18];
  const float* lnfs = (const float*)d_in[19];
  const float* lnfb = (const float*)d_in[20];
  const float* headW= (const float*)d_in[21];

  const size_t MD = (size_t)TOK * DIM;
  const size_t MF = (size_t)TOK * FFDIM;

  float* ws = (float*)d_ws;
  float* x  = ws;            // [TOK, D]
  float* h  = x  + MD;
  float* qb = h  + MD;
  float* kb2= qb + MD;
  float* vb = kb2+ MD;
  float* yb = vb + MD;
  float* z1 = yb + MD;       // [TOK, FF]
  float* z2 = qb;            // reuse q buffer post-attention
  const size_t fp32_floats = 6 * MD + MF;

  // bf16 region (only used on the full path)
  unsigned short* bb   = (unsigned short*)(ws + fp32_floats);
  unsigned short* h_bf = bb;
  unsigned short* y_bf = h_bf + MD;
  unsigned short* z1bf = y_bf + MD;
  unsigned short* zfbf = z1bf + MF;
  unsigned short* wqb  = zfbf + MD;
  const size_t WDD = (size_t)LAYERS * DIM * DIM;
  const size_t WFD = (size_t)LAYERS * FFDIM * DIM;
  unsigned short* wkb  = wqb + WDD;
  unsigned short* wvb  = wkb + WDD;
  unsigned short* wpb  = wvb + WDD;
  unsigned short* w1b  = wpb + WDD;
  unsigned short* w2b  = w1b + WFD;
  unsigned short* hwb  = w2b + WFD;
  const size_t bf16_ush = 3 * MD + MF + 4 * WDD + 2 * WFD + (size_t)VOCAB * DIM;
  const bool full = ws_size >= fp32_floats * 4 + bf16_ush * 2 + 64;

  float* logits = (float*)d_out;                 // [TOK, V]
  float* zf     = logits + (size_t)TOK * VOCAB;  // [TOK, D] (2nd tuple output)

  embed_kernel<<<TOK, 256, 0, stream>>>(idx, tok, pos, x);

  dim3 blk(256);
  dim3 gD(TOK / MBLK, DIM / NBLK);
  dim3 gFF(TOK / MBLK, FFDIM / NBLK);
  dim3 gV(TOK / MBLK, (VOCAB + NBLK - 1) / NBLK);

  if (full) {
    // one-time weight conversion into L2-resident bf16 cache
    auto cvt = [&](const float* src, unsigned short* dst, size_t n) {
      int n4 = (int)(n / 4);
      cvt_f32_to_bf16<<<(n4 + 255) / 256, 256, 0, stream>>>(src, dst, n4);
    };
    cvt(Wq, wqb, WDD); cvt(Wk, wkb, WDD); cvt(Wv, wvb, WDD); cvt(Wp, wpb, WDD);
    cvt(W1, w1b, WFD); cvt(W2, w2b, WFD);
    cvt(headW, hwb, (size_t)VOCAB * DIM);

    for (int l = 0; l < LAYERS; ++l) {
      layernorm_kernel<<<TOK, 256, 0, stream>>>(x, ln1s + l * DIM, ln1b + l * DIM, h, h_bf);
      gemm_bf16x2_wmma<<<gD, blk, 0, stream>>>(h_bf, wqb + l * (size_t)DIM * DIM,
                                               bq + l * DIM, nullptr, qb, TOK, DIM, DIM);
      gemm_bf16x2_wmma<<<gD, blk, 0, stream>>>(h_bf, wkb + l * (size_t)DIM * DIM,
                                               bk + l * DIM, nullptr, kb2, TOK, DIM, DIM);
      gemm_bf16x2_wmma<<<gD, blk, 0, stream>>>(h_bf, wvb + l * (size_t)DIM * DIM,
                                               bv + l * DIM, nullptr, vb, TOK, DIM, DIM);
      attention_kernel<<<BATCH * HEADS * TSEQ, 32, 0, stream>>>(qb, kb2, vb, yb, y_bf);
      gemm_bf16x2_wmma<<<gD, blk, 0, stream>>>(y_bf, wpb + l * (size_t)DIM * DIM,
                                               bp + l * DIM, x, x, TOK, DIM, DIM);
      layernorm_kernel<<<TOK, 256, 0, stream>>>(x, ln2s + l * DIM, ln2b + l * DIM, h, h_bf);
      gemm_bf16x2_wmma<<<gFF, blk, 0, stream>>>(h_bf, w1b + l * (size_t)FFDIM * DIM,
                                                b1 + l * FFDIM, nullptr, z1, TOK, FFDIM, DIM);
      kwinners_kernel<<<TOK, 256, 0, stream>>>(z1, FFDIM, KW1, 1, z1bf);
      gemm_bf16x2_wmma<<<gD, blk, 0, stream>>>(z1bf, w2b + l * (size_t)DIM * FFDIM,
                                               b2 + l * DIM, nullptr, z2, TOK, DIM, FFDIM);
      kwinners_kernel<<<TOK, 256, 0, stream>>>(z2, DIM, KW2, 0, nullptr);
      add_kernel<<<(TOK * DIM + 255) / 256, 256, 0, stream>>>(x, z2, TOK * DIM);
    }
    layernorm_kernel<<<TOK, 256, 0, stream>>>(x, lnfs, lnfb, zf, zfbf);
    gemm_bf16x2_wmma<<<gV, blk, 0, stream>>>(zfbf, hwb, nullptr, nullptr, logits,
                                             TOK, VOCAB, DIM);
  } else {
    // fallback: fp32-input GEMM with in-kernel conversion (small workspace)
    for (int l = 0; l < LAYERS; ++l) {
      layernorm_kernel<<<TOK, 256, 0, stream>>>(x, ln1s + l * DIM, ln1b + l * DIM, h, nullptr);
      gemm_bf16_wmma<<<gD, blk, 0, stream>>>(h, Wq + (size_t)l * DIM * DIM, bq + l * DIM,
                                             nullptr, qb, TOK, DIM, DIM);
      gemm_bf16_wmma<<<gD, blk, 0, stream>>>(h, Wk + (size_t)l * DIM * DIM, bk + l * DIM,
                                             nullptr, kb2, TOK, DIM, DIM);
      gemm_bf16_wmma<<<gD, blk, 0, stream>>>(h, Wv + (size_t)l * DIM * DIM, bv + l * DIM,
                                             nullptr, vb, TOK, DIM, DIM);
      attention_kernel<<<BATCH * HEADS * TSEQ, 32, 0, stream>>>(qb, kb2, vb, yb, nullptr);
      gemm_bf16_wmma<<<gD, blk, 0, stream>>>(yb, Wp + (size_t)l * DIM * DIM, bp + l * DIM,
                                             x, x, TOK, DIM, DIM);
      layernorm_kernel<<<TOK, 256, 0, stream>>>(x, ln2s + l * DIM, ln2b + l * DIM, h, nullptr);
      gemm_bf16_wmma<<<gFF, blk, 0, stream>>>(h, W1 + (size_t)l * FFDIM * DIM, b1 + l * FFDIM,
                                              nullptr, z1, TOK, FFDIM, DIM);
      kwinners_kernel<<<TOK, 256, 0, stream>>>(z1, FFDIM, KW1, 1, nullptr);
      gemm_bf16_wmma<<<gD, blk, 0, stream>>>(z1, W2 + (size_t)l * DIM * FFDIM, b2 + l * DIM,
                                             nullptr, z2, TOK, DIM, FFDIM);
      kwinners_kernel<<<TOK, 256, 0, stream>>>(z2, DIM, KW2, 0, nullptr);
      add_kernel<<<(TOK * DIM + 255) / 256, 256, 0, stream>>>(x, z2, TOK * DIM);
    }
    layernorm_kernel<<<TOK, 256, 0, stream>>>(x, lnfs, lnfb, zf, nullptr);
    gemm_bf16_wmma<<<gV, blk, 0, stream>>>(zf, headW, nullptr, nullptr, logits,
                                           TOK, VOCAB, DIM);
  }
}
